// GeneralizedAttention_19920058319249
// MI455X (gfx1250) — matrix operand order repaired and checked by reading, not compiled, not run on hardware
//
#include <hip/hip_runtime.h>
#include <hip/hip_bf16.h>
#include <math.h>

// Problem constants (match reference)
#define Bc   4
#define Hc   16
#define Nc   4096
#define Dc   64
#define Mc   256
#define BHc  (Bc * Hc)
#define EPSc 1e-6f
#define CHUNKS 16          // phase1 blocks per head; each covers Nc/CHUNKS = 256 rows

typedef __attribute__((ext_vector_type(16))) __bf16       v16bf;
typedef __attribute__((ext_vector_type(8)))  float        v8f;
typedef __attribute__((ext_vector_type(4)))  unsigned int u32x4;
typedef __attribute__((ext_vector_type(4)))  float        f32x4;

union FragBF { v16bf v; u32x4 q[2]; __bf16 h[16]; };
union PackBF { unsigned u; __bf16 h[2]; };

static __device__ __forceinline__ float gelu_erf(float x) {
    return 0.5f * x * (1.0f + erff(x * 0.70710678118654752440f));
}

static __device__ __forceinline__ v8f wmma_bf16(v16bf a, v16bf b, v8f c) {
    // D(16x16,f32) = A(16x32,bf16) * B(32x16,bf16) + C
    return __builtin_amdgcn_wmma_f32_16x16x32_bf16(false, a, false, b, (short)0, c, false, false);
}

// A operand 16x32 bf16. Source row-major; rowptr = start of this lane's row.
// ISA layout: lanes 0-15 hold K 0..7 (VGPR0-3) and K 16..23 (VGPR4-7);
// lanes 16-31 hold K 8..15 and K 24..31.
static __device__ __forceinline__ v16bf load_A_bf16(const __bf16* rowptr, int kbase, int lane) {
    const int kh = (lane & 16) ? 8 : 0;
    FragBF f;
    f.q[0] = *(const u32x4*)(rowptr + kbase + kh);
    f.q[1] = *(const u32x4*)(rowptr + kbase + kh + 16);
    return f.v;
}

// Same as above but converting from f32 source (q/k rows in global memory).
static __device__ __forceinline__ v16bf load_A_f32(const float* rowptr, int kbase, int lane) {
    const int kh = (lane & 16) ? 8 : 0;
    const float* p0 = rowptr + kbase + kh;
    const float* p1 = p0 + 16;
    FragBF f;
#pragma unroll
    for (int i = 0; i < 8; ++i) { f.h[i] = (__bf16)p0[i]; f.h[8 + i] = (__bf16)p1[i]; }
    return f.v;
}

// B operand 32x16 bf16. Source stored col-major [col][K], K contiguous;
// colptr = start of this lane's column. Lanes 0-15: K 0..7, 16..23; lanes 16-31: K 8..15, 24..31.
static __device__ __forceinline__ v16bf load_B_bf16(const __bf16* colptr, int kbase, int lane) {
    const int kh = (lane & 16) ? 8 : 0;
    FragBF f;
    f.q[0] = *(const u32x4*)(colptr + kbase + kh);
    f.q[1] = *(const u32x4*)(colptr + kbase + kh + 16);
    return f.v;
}

// ---------------------------------------------------------------------------
// Kernel 0: zero the contexts/ksum accumulators; convert proj_mat -> bf16.
// ---------------------------------------------------------------------------
__global__ __launch_bounds__(256) void ga_init(
    const float* __restrict__ proj, float* __restrict__ ctx_ws,
    float* __restrict__ ksum_ws, __bf16* __restrict__ projb)
{
    const int zero_ctx  = BHc * Mc * Dc;
    const int zero_all  = zero_ctx + BHc * Mc;
    const int total     = zero_all + Mc * Dc;
    for (int i = blockIdx.x * blockDim.x + threadIdx.x; i < total;
         i += gridDim.x * blockDim.x) {
        if (i < zero_ctx)       ctx_ws[i] = 0.0f;
        else if (i < zero_all)  ksum_ws[i - zero_ctx] = 0.0f;
        else                    projb[i - zero_all] = (__bf16)proj[i - zero_all];
    }
}

// ---------------------------------------------------------------------------
// Kernel 1: per (head, chunk): kp = gelu(k·projT)+eps (WMMA), accumulate
// k_sum (LDS ds_add_f32) and contexts += kp^T·v (WMMA, LDS transpose staging,
// global f32 atomics into L2-resident workspace).
// ---------------------------------------------------------------------------
__global__ __launch_bounds__(256) void ga_phase1(
    const float* __restrict__ kin, const float* __restrict__ vin,
    const __bf16* __restrict__ projb,
    float* __restrict__ ctx_ws, float* __restrict__ ksum_ws)
{
    __shared__ __attribute__((aligned(16))) __bf16 lds_kpT[Mc * 32]; // [m][n] (A of GEMM2)
    __shared__ __attribute__((aligned(16))) __bf16 lds_vT[Dc * 32];  // [e][n] (B of GEMM2)
    __shared__ float lds_ksum[Mc];

    const int tid  = threadIdx.x, wave = tid >> 5, lane = tid & 31;
    const int bh    = blockIdx.x / CHUNKS;
    const int chunk = blockIdx.x % CHUNKS;
    const float* kg = kin + (size_t)bh * Nc * Dc;
    const float* vg = vin + (size_t)bh * Nc * Dc;

    lds_ksum[tid] = 0.0f;

    v8f acc[8];                            // wave-local contexts: 2 m-tiles x 4 e-tiles
#pragma unroll
    for (int i = 0; i < 8; ++i) acc[i] = (v8f){};

    const int n0 = chunk * (Nc / CHUNKS);
    for (int stripe = 0; stripe < (Nc / CHUNKS) / 32; ++stripe) {
        const int nb = n0 + stripe * 32;
        __syncthreads();                   // previous stripe fully consumed

        // stage v stripe transposed into LDS as bf16 [e][n]:
        // thread -> n-pair (2*n2, 2*n2+1) x 4 consecutive e; b128 global loads,
        // bf16-pair packed b32 LDS stores (n-adjacent elements are contiguous).
        {
            const int n2 = tid >> 4;            // 0..15 -> rows 2*n2, 2*n2+1
            const int eo = (tid & 15) * 4;      // 4 consecutive e
            const float* vrow0 = vg + (size_t)(nb + 2 * n2) * Dc + eo;
            const f32x4 a0 = *(const f32x4*)vrow0;
            const f32x4 a1 = *(const f32x4*)(vrow0 + Dc);
#pragma unroll
            for (int j = 0; j < 4; ++j) {
                PackBF p; p.h[0] = (__bf16)a0[j]; p.h[1] = (__bf16)a1[j];
                *(unsigned*)&lds_vT[(eo + j) * 32 + 2 * n2] = p.u;
            }
        }

        // GEMM1: kp tile (32n x 256m) -> 32 tiles of 16x16, 4 per wave
#pragma unroll
        for (int t = 0; t < 4; ++t) {
            const int tile = wave * 4 + t;
            const int nsub = tile >> 4, mt = tile & 15;
            const float*  arow = kg + (size_t)(nb + nsub * 16 + (lane & 15)) * Dc;
            const __bf16* bcol = projb + (size_t)(mt * 16 + (lane & 15)) * Dc;
            v8f c = {};
            c = wmma_bf16(load_A_f32(arow,  0, lane), load_B_bf16(bcol,  0, lane), c);
            c = wmma_bf16(load_A_f32(arow, 32, lane), load_B_bf16(bcol, 32, lane), c);
            const int mcol  = mt * 16 + (lane & 15);
            const int rbase = nsub * 16 + ((lane & 16) ? 8 : 0);
            float ksum_part = 0.0f;
#pragma unroll
            for (int r = 0; r < 8; r += 2) {   // pack row pairs -> b32 LDS stores
                const float v0 = gelu_erf(c[r])     + EPSc;
                const float v1 = gelu_erf(c[r + 1]) + EPSc;
                ksum_part += v0 + v1;
                PackBF p; p.h[0] = (__bf16)v0; p.h[1] = (__bf16)v1;
                *(unsigned*)&lds_kpT[mcol * 32 + rbase + r] = p.u;
            }
            atomicAdd(&lds_ksum[mcol], ksum_part);
        }
        __syncthreads();

        // GEMM2: contexts(m,e) += kp^T(m,n) * v(n,e), single K=32 WMMA per tile
#pragma unroll
        for (int mt2 = 0; mt2 < 2; ++mt2) {
            const int mrow = (wave * 2 + mt2) * 16 + (lane & 15);
            const v16bf a = load_A_bf16(lds_kpT + mrow * 32, 0, lane);
#pragma unroll
            for (int et = 0; et < 4; ++et) {
                const v16bf b = load_B_bf16(lds_vT + (et * 16 + (lane & 15)) * 32, 0, lane);
                acc[mt2 * 4 + et] = wmma_bf16(a, b, acc[mt2 * 4 + et]);
            }
        }
    }
    __syncthreads();

    // epilogue: accumulate into workspace (stays resident in 192MB L2)
    float* ctx = ctx_ws + (size_t)bh * Mc * Dc;
#pragma unroll
    for (int mt2 = 0; mt2 < 2; ++mt2) {
        const int mb = (wave * 2 + mt2) * 16 + ((lane & 16) ? 8 : 0);
#pragma unroll
        for (int et = 0; et < 4; ++et) {
            const int e = et * 16 + (lane & 15);
#pragma unroll
            for (int r = 0; r < 8; ++r)
                atomicAdd(&ctx[(size_t)(mb + r) * Dc + e], acc[mt2 * 4 + et][r]);
        }
    }
    atomicAdd(&ksum_ws[bh * Mc + tid], lds_ksum[tid]);
}

// ---------------------------------------------------------------------------
// Kernel 1.5: one-time contexts f32 [m][e] -> bf16 transposed [e][m] so
// phase2 can stage it with pure async b128 copies (no per-block conversion).
// ---------------------------------------------------------------------------
__global__ __launch_bounds__(256) void ga_finalize(
    const float* __restrict__ ctx_ws, __bf16* __restrict__ ctxT_ws)
{
    const int bh = blockIdx.x;
    const float* ctx = ctx_ws + (size_t)bh * Mc * Dc;
    __bf16* dst = ctxT_ws + (size_t)bh * Dc * Mc;
    for (int i = threadIdx.x; i < Mc * Dc; i += 256) {
        const int e = i >> 8, m = i & 255;               // dst[e][m], coalesced writes
        dst[i] = (__bf16)ctx[(size_t)m * Dc + e];
    }
}

// ---------------------------------------------------------------------------
// Kernel 2: per (head, 64-row stripe): qp = gelu(q·projT)+eps (WMMA),
// d_inv = 1/(qp·k_sum), out = (qp·contexts) * d_inv (WMMA).
// contexts^T staged global->LDS with GLOBAL_LOAD_ASYNC_TO_LDS_B128 (ASYNCcnt),
// overlapped with the qp WMMA work.
// ---------------------------------------------------------------------------
__global__ __launch_bounds__(256) void ga_phase2(
    const float* __restrict__ qin, const __bf16* __restrict__ projb,
    const __bf16* __restrict__ ctxT_ws, const float* __restrict__ ksum_ws,
    float* __restrict__ out)
{
    __shared__ __attribute__((aligned(16))) __bf16 lds_qp[64 * Mc];   // [n][m] (A of GEMM3)
    __shared__ __attribute__((aligned(16))) __bf16 lds_ctxT[Dc * Mc]; // [e][m] (B of GEMM3)
    __shared__ float lds_ksum[Mc];
    __shared__ float lds_den[64];
    __shared__ float lds_dinv[64];

    const int tid = threadIdx.x, wave = tid >> 5, lane = tid & 31;
    const int bh = blockIdx.x >> 6;        // 64 stripes of 64 rows per head
    const int nb = (blockIdx.x & 63) * 64;
    const float*  qg   = qin     + (size_t)bh * Nc * Dc;
    const __bf16* ctxT = ctxT_ws + (size_t)bh * Dc * Mc;

    lds_ksum[tid] = ksum_ws[bh * Mc + tid];
    if (tid < 64) lds_den[tid] = 0.0f;

    // async-stage contexts^T (bf16, 32KB) global -> LDS; 16B per lane per op.
    // GVS mode: saddr = uniform 64-bit base (SGPR pair), vaddr = 32-bit offset.
    {
        const unsigned long long gbase = (unsigned long long)ctxT;
#pragma unroll
        for (int it = 0; it < (Dc * Mc) / (256 * 8); ++it) {
            const int i = it * 256 + tid;                 // i-th 16B chunk
            const unsigned lds_off = (unsigned)(unsigned long long)(const void*)&lds_ctxT[i * 8];
            const unsigned goff    = (unsigned)(i * 16);
            asm volatile("global_load_async_to_lds_b128 %0, %1, %2"
                         :: "v"(lds_off), "v"(goff), "s"(gbase)
                         : "memory");
        }
    }

    // GEMM1: qp tile (64n x 256m) -> 64 tiles of 16x16, 8 per wave
    // (overlaps the async contexts^T copy above)
#pragma unroll
    for (int t = 0; t < 8; ++t) {
        const int tile = wave * 8 + t;
        const int nsub = tile >> 4, mt = tile & 15;
        const float*  arow = qg + (size_t)(nb + nsub * 16 + (lane & 15)) * Dc;
        const __bf16* bcol = projb + (size_t)(mt * 16 + (lane & 15)) * Dc;
        v8f c = {};
        c = wmma_bf16(load_A_f32(arow,  0, lane), load_B_bf16(bcol,  0, lane), c);
        c = wmma_bf16(load_A_f32(arow, 32, lane), load_B_bf16(bcol, 32, lane), c);
        const int mcol  = mt * 16 + (lane & 15);
        const int rbase = nsub * 16 + ((lane & 16) ? 8 : 0);
#pragma unroll
        for (int r = 0; r < 8; ++r)
            lds_qp[(rbase + r) * Mc + mcol] = (__bf16)(gelu_erf(c[r]) + EPSc);
    }
    asm volatile("s_wait_asynccnt 0x0" ::: "memory");  // ctx^T copy landed in LDS
    __syncthreads();

    // denom[n] = sum_m qp[n,m]*ksum[m]; 4 threads per row + ds_add_f32 reduce
    {
        const int n = tid >> 2, part = (tid & 3) * 64;
        float s = 0.0f;
        for (int m = part; m < part + 64; ++m)
            s += (float)lds_qp[n * Mc + m] * lds_ksum[m];
        atomicAdd(&lds_den[n], s);
    }
    __syncthreads();
    if (tid < 64) lds_dinv[tid] = 1.0f / lds_den[tid];
    __syncthreads();

    // GEMM3: out(n,e) = qp(n,m) * contexts(m,e); 16 tiles -> 2 per wave, K=256
#pragma unroll
    for (int t2 = 0; t2 < 2; ++t2) {
        const int tile = wave * 2 + t2;
        const int nsub = tile >> 2, et = tile & 3;
        const __bf16* arow = lds_qp   + (nsub * 16 + (lane & 15)) * Mc;
        const __bf16* bcol = lds_ctxT + (et   * 16 + (lane & 15)) * Mc;
        v8f c = {};
#pragma unroll
        for (int ks = 0; ks < 8; ++ks)
            c = wmma_bf16(load_A_bf16(arow, ks * 32, lane),
                          load_B_bf16(bcol, ks * 32, lane), c);
        const int e  = et * 16 + (lane & 15);
        const int rb = nsub * 16 + ((lane & 16) ? 8 : 0);
#pragma unroll
        for (int r = 0; r < 8; ++r) {
            const int n = rb + r;
            out[((size_t)bh * Nc + nb + n) * Dc + e] = c[r] * lds_dinv[n];
        }
    }
}

// ---------------------------------------------------------------------------
extern "C" void kernel_launch(void* const* d_in, const int* in_sizes, int n_in,
                              void* d_out, int out_size, void* d_ws, size_t ws_size,
                              hipStream_t stream) {
    (void)in_sizes; (void)n_in; (void)out_size; (void)ws_size;
    const float* q    = (const float*)d_in[0];
    const float* k    = (const float*)d_in[1];
    const float* v    = (const float*)d_in[2];
    const float* proj = (const float*)d_in[3];
    float* out = (float*)d_out;

    // workspace layout:
    // [contexts f32: BH*M*D][ksum f32: BH*M][proj bf16: M*D][ctx^T bf16: BH*D*M]
    float*  ctx_ws  = (float*)d_ws;
    float*  ksum_ws = ctx_ws + (size_t)BHc * Mc * Dc;
    __bf16* projb   = (__bf16*)(ksum_ws + (size_t)BHc * Mc);
    __bf16* ctxT_ws = projb + (size_t)Mc * Dc;

    ga_init    <<<512,             256, 0, stream>>>(proj, ctx_ws, ksum_ws, projb);
    ga_phase1  <<<BHc * CHUNKS,    256, 0, stream>>>(k, v, projb, ctx_ws, ksum_ws);
    ga_finalize<<<BHc,             256, 0, stream>>>(ctx_ws, ctxT_ws);
    ga_phase2  <<<BHc * (Nc / 64), 256, 0, stream>>>(q, projb, ctxT_ws, ksum_ws, out);
}
